// FCGRU_47983374631290
// MI455X (gfx1250) — matrix operand order repaired and checked by reading, not compiled, not run on hardware
//
#include <hip/hip_runtime.h>
#include <math.h>

// ---------------- problem constants ----------------
#define BB    1024   // batch
#define TT    128    // time steps
#define DIN   32
#define HH    512
#define D2    256
#define DOUT  16
#define GC    2048   // packed gate columns: [rz 1024 | xn 512 | hnn 512]
#define LW    20     // LDS row stride in dwords: 16 data + 4 pad (16B-aligned, bank-clean)

typedef unsigned int u32;
typedef unsigned short u16;
typedef unsigned long long u64;
typedef __attribute__((ext_vector_type(8)))  float  v8f;
typedef __attribute__((ext_vector_type(16))) __bf16 v16bf;
typedef __attribute__((ext_vector_type(4)))  int    v4i;
typedef __attribute__((ext_vector_type(8)))  int    v8i;

union FragU { u32 u[8]; v16bf v; };

__device__ __forceinline__ u16 f2bf(float f) {
    u32 u = __float_as_uint(f);
    u += 0x7FFFu + ((u >> 16) & 1u);     // round-to-nearest-even
    return (u16)(u >> 16);
}
__device__ __forceinline__ u32 pack2(u16 lo, u16 hi) {
    return (u32)lo | ((u32)hi << 16);
}

// ---------------- TDM: DMA a (tile_d0 x tile_d1) bf16 tile into LDS ----------------
// 2-D tensor, data_size=2B, LDS padding: 4 dwords after every 16 dwords -> LW=20 rows.
// D# group layouts per CDNA5 ISA 8.3/8.4.
__device__ __forceinline__ void tdm_load_tile(u32 lds_off, const void* gptr,
                                              u32 tensor_d0, u32 tensor_d1,
                                              u32 tile_d0, u32 tile_d1,
                                              u32 stride_elems) {
    u64 ga = (u64)(size_t)gptr;
    v4i g0;
    g0[0] = 1;                                         // count=1, user mode
    g0[1] = (int)lds_off;                              // lds_addr (bytes)
    g0[2] = (int)(u32)ga;                              // global_addr[31:0]
    g0[3] = (int)(((u32)(ga >> 32)) & 0x01FFFFFFu) | (2 << 30);  // addr[56:32] | type=2
    v8i g1;
    g1[0] = (1 << 16)      // data_size = 2 bytes
          | (1 << 20)      // pad_enable
          | (3 << 22)      // pad_interval: 16 dwords
          | (3 << 25);     // pad_amount: 4 dwords
    g1[1] = (int)((tensor_d0 & 0xFFFFu) << 16);                     // dim0[15:0]
    g1[2] = (int)(((tensor_d0 >> 16) & 0xFFFFu) | ((tensor_d1 & 0xFFFFu) << 16));
    g1[3] = (int)(((tensor_d1 >> 16) & 0xFFFFu) | (tile_d0 << 16)); // tile_dim0
    g1[4] = (int)(tile_d1 & 0xFFFFu);                               // tile_dim1 (tile_dim2=0)
    g1[5] = (int)stride_elems;                                      // dim0_stride[31:0]
    g1[6] = 0;                                                      // stride hi / dim1_stride lo
    g1[7] = 0;
    asm volatile("tensor_load_to_lds %0, %1" :: "s"(g0), "s"(g1) : "memory");
}

// ------- shared WMMA micro-kernel: one BK=32 chunk, 8 waves, 2x4 wave grid -------
// Each wave: 4 M-subtiles x 2 N-subtiles of 16x16, bf16 A/B, f32 accum.
__device__ __forceinline__ void wmma_chunk(const u32 (*As)[LW], const u32 (*Bs)[LW],
                                           int lane, int wm, int wn, v8f acc[4][2]) {
    FragU a[4], b[2];
    const int ml    = lane & 15;
    const int half  = lane >> 4;      // 0 or 1
    const int khalf = half * 4;       // u32-column base for A
#pragma unroll
    for (int mt = 0; mt < 4; ++mt) {
        int m = wm + mt * 16 + ml;
        uint4 q0 = *(const uint4*)&As[m][khalf];
        uint4 q1 = *(const uint4*)&As[m][8 + khalf];
        a[mt].u[0] = q0.x; a[mt].u[1] = q0.y; a[mt].u[2] = q0.z; a[mt].u[3] = q0.w;
        a[mt].u[4] = q1.x; a[mt].u[5] = q1.y; a[mt].u[6] = q1.z; a[mt].u[7] = q1.w;
    }
#pragma unroll
    for (int nt = 0; nt < 2; ++nt) {
        int n = wn + nt * 16 + ml;
        uint4 q0 = *(const uint4*)&Bs[n][half * 4];
        uint4 q1 = *(const uint4*)&Bs[n][8 + half * 4];
        b[nt].u[0] = q0.x; b[nt].u[1] = q0.y; b[nt].u[2] = q0.z; b[nt].u[3] = q0.w;
        b[nt].u[4] = q1.x; b[nt].u[5] = q1.y; b[nt].u[6] = q1.z; b[nt].u[7] = q1.w;
    }
#pragma unroll
    for (int mt = 0; mt < 4; ++mt)
#pragma unroll
        for (int nt = 0; nt < 2; ++nt)
            acc[mt][nt] = __builtin_amdgcn_wmma_f32_16x16x32_bf16(
                false, a[mt].v, false, b[nt].v, (short)0, acc[mt][nt], false, false);
}

__device__ __forceinline__ void acc_zero(v8f acc[4][2]) {
#pragma unroll
    for (int mt = 0; mt < 4; ++mt)
#pragma unroll
        for (int nt = 0; nt < 2; ++nt) {
            v8f z = {0.f,0.f,0.f,0.f,0.f,0.f,0.f,0.f};
            acc[mt][nt] = z;
        }
}

// ---------------- prep kernels ----------------
// W12 = W2 @ W1 (512x32) bf16 ; b12 = W2@b1 + b2
__global__ void k_prep_w12(const float* __restrict__ W2, const float* __restrict__ W1,
                           const float* __restrict__ b1, const float* __restrict__ b2,
                           u16* __restrict__ W12b, float* __restrict__ b12) {
    int idx = blockIdx.x * 256 + threadIdx.x;
    if (idx < 512 * 32) {
        int n = idx >> 5, k = idx & 31;
        float s = 0.f;
        for (int j = 0; j < 256; ++j) s += W2[n * 256 + j] * W1[j * 32 + k];
        W12b[idx] = f2bf(s);
    } else if (idx < 512 * 32 + 512) {
        int n = idx - 512 * 32;
        float s = b2[n];
        for (int j = 0; j < 256; ++j) s += W2[n * 256 + j] * b1[j];
        b12[n] = s;
    }
}

// Packed recurrent weights: Wpack[n][k], n<2048, k<1024 (bf16, k-contiguous)
__global__ void k_prep_pack(const float* __restrict__ W_ih, const float* __restrict__ W_hh,
                            u16* __restrict__ Wpackb) {
    int idx = blockIdx.x * 256 + threadIdx.x;           // 2048*1024 total
    int n = idx >> 10, k = idx & 1023;
    float v;
    if (n < 1024)       v = (k < 512) ? W_ih[n * 512 + k] : W_hh[n * 512 + (k - 512)];
    else if (n < 1536)  v = (k < 512) ? W_ih[n * 512 + k] : 0.f;     // xn rows 1024..1535
    else                v = (k >= 512) ? W_hh[(n - 512) * 512 + (k - 512)] : 0.f; // hnn
    Wpackb[idx] = f2bf(v);
}

// h / hbf init, W3 -> bf16, packed bias
__global__ void k_prep_misc(const float* __restrict__ hn, const float* __restrict__ W3,
                            const float* __restrict__ b_ih, const float* __restrict__ b_hh,
                            float* __restrict__ h, u16* __restrict__ hbf,
                            u16* __restrict__ W3b, float* __restrict__ bpack) {
    int idx = blockIdx.x * 256 + threadIdx.x;           // 524288 threads
    float hv = hn[idx];
    h[idx]   = hv;
    hbf[idx] = f2bf(hv);
    if (idx < 256 * 512) W3b[idx] = f2bf(W3[idx]);
    if (idx < GC) {
        float v;
        if (idx < 1024)      v = b_ih[idx] + b_hh[idx];
        else if (idx < 1536) v = b_ih[idx];
        else                 v = b_hh[idx - 512];
        bpack[idx] = v;
    }
}

// ---------------- X = input @ W12^T + b12  (rows r = t*1024+b), bf16 out ----------------
__global__ __launch_bounds__(256) void k_gemm_x(const float* __restrict__ input,
                                                const u16* __restrict__ W12b,
                                                const float* __restrict__ b12,
                                                u16* __restrict__ X) {
    __shared__ __align__(16) u32 As[128][LW];
    __shared__ __align__(16) u32 Bs[128][LW];
    const int tid = threadIdx.x;
    const int r0 = blockIdx.y * 128;
    const int n0 = blockIdx.x * 128;
    for (int i = tid; i < 128 * 16; i += 256) {
        int row = i >> 4, col = i & 15;
        int r = r0 + row, b = r & 1023, t = r >> 10;
        const float* src = input + ((size_t)b * TT + t) * DIN + col * 2;
        As[row][col] = pack2(f2bf(src[0]), f2bf(src[1]));
    }
    const u32* Wu = (const u32*)W12b;                   // row stride 16 u32
    for (int i = tid; i < 128 * 16; i += 256) {
        int row = i >> 4, col = i & 15;
        Bs[row][col] = Wu[(size_t)(n0 + row) * 16 + col];
    }
    __syncthreads();

    const int lane = tid & 31, wave = tid >> 5;
    const int wm = (wave & 1) * 64, wn = (wave >> 1) * 32;
    v8f acc[4][2];
    acc_zero(acc);
    wmma_chunk(As, Bs, lane, wm, wn, acc);              // K=32 in one shot

    const int ml = lane & 15, half = lane >> 4;
#pragma unroll
    for (int mt = 0; mt < 4; ++mt)
#pragma unroll
        for (int nt = 0; nt < 2; ++nt)
#pragma unroll
            for (int rr = 0; rr < 8; ++rr) {
                int m = wm + mt * 16 + rr + half * 8;
                int n = wn + nt * 16 + ml;
                float v = acc[mt][nt][rr] + b12[n0 + n];
                X[(size_t)(r0 + m) * HH + (n0 + n)] = f2bf(v);
            }
}

// ---- per-step gate GEMM: G = [x_t|h] @ Wpack^T + bpack, TDM-staged, double-buffered ----
__global__ __launch_bounds__(256) void k_gemm_step(const u16* __restrict__ X,
                                                   const u16* __restrict__ hbf,
                                                   const u16* __restrict__ Wpackb,
                                                   const float* __restrict__ bpack,
                                                   float* __restrict__ G, int t) {
    __shared__ __align__(16) u32 As[2][128][LW];
    __shared__ __align__(16) u32 Bs[2][128][LW];
    const int tid = threadIdx.x;
    const int c0 = blockIdx.x * 128;                    // packed gate column block
    const int b0 = blockIdx.y * 128;                    // batch block
    int ks, ke;
    if (c0 < 1024)      { ks = 0;   ke = 1024; }        // r,z: full [x|h]
    else if (c0 < 1536) { ks = 0;   ke = 512;  }        // xn: x only
    else                { ks = 512; ke = 1024; }        // hnn: h only

    const int lane = tid & 31, wave = tid >> 5;
    const int wm = (wave & 1) * 64, wn = (wave >> 1) * 32;
    v8f acc[4][2];
    acc_zero(acc);

    const u32 ldsA0 = (u32)(size_t)&As[0][0][0];
    const u32 ldsA1 = (u32)(size_t)&As[1][0][0];
    const u32 ldsB0 = (u32)(size_t)&Bs[0][0][0];
    const u32 ldsB1 = (u32)(size_t)&Bs[1][0][0];

    // A-tile source for K-chunk kc: X rows (t*BB+b0..) for kc<512, hbf rows for kc>=512
    auto a_src = [&](int kc) -> const void* {
        return (kc < 512)
            ? (const void*)(X   + ((size_t)(t * BB + b0) * HH + kc))
            : (const void*)(hbf + ((size_t)b0 * HH + (kc - 512)));
    };

    if (wave == 0) {
        tdm_load_tile(ldsA0, a_src(ks), 512, BB * TT, 32, 128, HH);
        tdm_load_tile(ldsB0, Wpackb + ((size_t)c0 * 1024 + ks), 1024, GC, 32, 128, 1024);
        __builtin_amdgcn_s_wait_tensorcnt(0);
    }
    __syncthreads();

    int p = 0;
    for (int kc = ks; kc < ke; kc += 32) {
        int nkc = kc + 32;
        if (nkc < ke && wave == 0) {                    // prefetch next chunk via TDM
            tdm_load_tile(p ? ldsA0 : ldsA1, a_src(nkc), 512, BB * TT, 32, 128, HH);
            tdm_load_tile(p ? ldsB0 : ldsB1,
                          Wpackb + ((size_t)c0 * 1024 + nkc), 1024, GC, 32, 128, 1024);
        }
        wmma_chunk(As[p], Bs[p], lane, wm, wn, acc);
        if (wave == 0) __builtin_amdgcn_s_wait_tensorcnt(0);
        __syncthreads();
        p ^= 1;
    }

    const int ml = lane & 15, half = lane >> 4;
#pragma unroll
    for (int mt = 0; mt < 4; ++mt)
#pragma unroll
        for (int nt = 0; nt < 2; ++nt)
#pragma unroll
            for (int rr = 0; rr < 8; ++rr) {
                int m = wm + mt * 16 + rr + half * 8;
                int n = wn + nt * 16 + ml;
                G[(size_t)(b0 + m) * GC + (c0 + n)] = acc[mt][nt][rr] + bpack[c0 + n];
            }
}

// ---------------- gate elementwise + h update (keeps f32 + bf16 mirrors) ----------------
__global__ void k_gate(const float* __restrict__ G, float* __restrict__ h,
                       u16* __restrict__ hbf, float* __restrict__ out_hn, int write_out) {
    int idx = blockIdx.x * 256 + threadIdx.x;           // 1024*512
    int b = idx >> 9, j = idx & 511;
    size_t base = (size_t)b * GC;
    float r   = 1.f / (1.f + expf(-G[base + j]));
    float z   = 1.f / (1.f + expf(-G[base + 512 + j]));
    float xn  = G[base + 1024 + j];
    float hnn = G[base + 1536 + j];
    float nv  = tanhf(xn + r * hnn);
    float hv  = (1.f - z) * nv + z * h[idx];
    h[idx]   = hv;
    hbf[idx] = f2bf(hv);
    if (write_out) out_hn[idx] = hv;                    // hn_out = h_last
}

// ---------------- O = relu(relu(h) @ W3^T + b3)  (1024x256, K=512) ----------------
__global__ __launch_bounds__(256) void k_gemm_out(const float* __restrict__ h,
                                                  const u16* __restrict__ W3b,
                                                  const float* __restrict__ b3,
                                                  float* __restrict__ O) {
    __shared__ __align__(16) u32 As[128][LW];
    __shared__ __align__(16) u32 Bs[128][LW];
    const int tid = threadIdx.x;
    const int n0 = blockIdx.x * 128;
    const int b0 = blockIdx.y * 128;
    const int lane = tid & 31, wave = tid >> 5;
    const int wm = (wave & 1) * 64, wn = (wave >> 1) * 32;
    v8f acc[4][2];
    acc_zero(acc);
    const u32* Wu = (const u32*)W3b;                    // row stride 256 u32
    for (int kc = 0; kc < HH; kc += 32) {
        for (int i = tid; i < 128 * 16; i += 256) {
            int row = i >> 4, col = i & 15;
            const float* hp = h + (size_t)(b0 + row) * HH + kc + col * 2;
            As[row][col] = pack2(f2bf(fmaxf(hp[0], 0.f)), f2bf(fmaxf(hp[1], 0.f)));
        }
        for (int i = tid; i < 128 * 16; i += 256) {
            int row = i >> 4, col = i & 15;
            Bs[row][col] = Wu[(size_t)(n0 + row) * 256 + kc / 2 + col];
        }
        __syncthreads();
        wmma_chunk(As, Bs, lane, wm, wn, acc);
        __syncthreads();
    }
    const int ml = lane & 15, half = lane >> 4;
#pragma unroll
    for (int mt = 0; mt < 4; ++mt)
#pragma unroll
        for (int nt = 0; nt < 2; ++nt)
#pragma unroll
            for (int rr = 0; rr < 8; ++rr) {
                int m = wm + mt * 16 + rr + half * 8;
                int n = wn + nt * 16 + ml;
                O[(size_t)(b0 + m) * D2 + (n0 + n)] =
                    fmaxf(acc[mt][nt][rr] + b3[n0 + n], 0.f);
            }
}

// ---------------- params[b][o] = O[b] . Wh[cult[b]][o] + bh[cult[b]][o] ----------------
__global__ void k_params(const float* __restrict__ O, const int* __restrict__ cult,
                         const float* __restrict__ Wh, const float* __restrict__ bh,
                         float* __restrict__ out) {
    __shared__ float Os[256];
    const int b = blockIdx.x, tid = threadIdx.x;
    Os[tid] = O[(size_t)b * D2 + tid];
    __syncthreads();
    int c = cult[b];
    int o = tid >> 4, s = tid & 15;
    const float* w = Wh + ((size_t)c * DOUT + o) * D2;
    float p = 0.f;
    for (int d = s; d < D2; d += 16) p += Os[d] * w[d];
    p += __shfl_xor(p, 8, 32);
    p += __shfl_xor(p, 4, 32);
    p += __shfl_xor(p, 2, 32);
    p += __shfl_xor(p, 1, 32);
    if (s == 0) out[b * DOUT + o] = p + bh[c * DOUT + o];
}

// ---------------- launch ----------------
extern "C" void kernel_launch(void* const* d_in, const int* in_sizes, int n_in,
                              void* d_out, int out_size, void* d_ws, size_t ws_size,
                              hipStream_t stream) {
    const float* input = (const float*)d_in[0];
    const float* hn    = (const float*)d_in[1];
    const int*   cult  = (const int*)  d_in[2];
    const float* W1    = (const float*)d_in[3];
    const float* b1    = (const float*)d_in[4];
    const float* W2    = (const float*)d_in[5];
    const float* b2    = (const float*)d_in[6];
    const float* W_ih  = (const float*)d_in[7];
    const float* W_hh  = (const float*)d_in[8];
    const float* b_ih  = (const float*)d_in[9];
    const float* b_hh  = (const float*)d_in[10];
    const float* W3    = (const float*)d_in[11];
    const float* b3    = (const float*)d_in[12];
    const float* Wh    = (const float*)d_in[13];
    const float* bh    = (const float*)d_in[14];
    float* out = (float*)d_out;

    char* ws = (char*)d_ws;
    size_t off = 0;
    auto alloc = [&](size_t bytes) -> void* {
        void* p = ws + off;
        off = (off + bytes + 255) & ~(size_t)255;
        return p;
    };
    u16*   W12b   = (u16*)  alloc((size_t)512 * 32 * 2);
    float* b12    = (float*)alloc(512 * 4);
    u16*   Wpackb = (u16*)  alloc((size_t)GC * 1024 * 2);
    float* bpack  = (float*)alloc(GC * 4);
    u16*   W3b    = (u16*)  alloc((size_t)D2 * HH * 2);
    u16*   X      = (u16*)  alloc((size_t)TT * BB * HH * 2);
    float* h      = (float*)alloc((size_t)BB * HH * 4);
    u16*   hbf    = (u16*)  alloc((size_t)BB * HH * 2);
    float* G      = (float*)alloc((size_t)BB * GC * 4);
    float* O      = (float*)alloc((size_t)BB * D2 * 4);
    (void)in_sizes; (void)n_in; (void)out_size; (void)ws_size;

    k_prep_w12 <<<66,   256, 0, stream>>>(W2, W1, b1, b2, W12b, b12);
    k_prep_pack<<<8192, 256, 0, stream>>>(W_ih, W_hh, Wpackb);
    k_prep_misc<<<2048, 256, 0, stream>>>(hn, W3, b_ih, b_hh, h, hbf, W3b, bpack);

    k_gemm_x<<<dim3(4, 1024), 256, 0, stream>>>(input, W12b, b12, X);

    for (int t = 0; t < TT; ++t) {
        k_gemm_step<<<dim3(16, 8), 256, 0, stream>>>(X, hbf, Wpackb, bpack, G, t);
        k_gate<<<2048, 256, 0, stream>>>(G, h, hbf, out + BB * DOUT, (t == TT - 1) ? 1 : 0);
    }

    k_gemm_out<<<dim3(2, 8), 256, 0, stream>>>(h, W3b, b3, O);
    k_params  <<<1024, 256, 0, stream>>>(O, cult, Wh, bh, out);
}
